// GNN_24618752540869
// MI455X (gfx1250) — matrix operand order repaired and checked by reading, not compiled, not run on hardware
//
#include <hip/hip_runtime.h>

typedef __attribute__((ext_vector_type(2)))  float    v2f;
typedef __attribute__((ext_vector_type(8)))  float    v8f;
typedef __attribute__((ext_vector_type(16))) __bf16   v16bf;
typedef __attribute__((ext_vector_type(4)))  unsigned v4u;

struct ABpack { v4u a, b; };   // 32 bytes -> bit_cast to v16bf

#define BN 4
#define NN 4096
#define CC 64
#define THRESH 0.005f
#define EPSN 1e-5f

static __device__ __forceinline__ v8f wmma_f32(v2f a, v2f b, v8f c) {
    return __builtin_amdgcn_wmma_f32_16x16x4_f32(false, a, false, b, (short)0, c,
                                                 false, false);
}

static __device__ __forceinline__ unsigned short f2bf(float f) {
    unsigned u = __builtin_bit_cast(unsigned, f);
    unsigned r = (u + 0x7FFFu + ((u >> 16) & 1u)) >> 16;   // RNE
    return (unsigned short)r;
}

// ---------------------------------------------------------------------------
// 1) x [B,C,N] -> xcur [B,N,C]
__global__ void k_transpose(const float* __restrict__ x, float* __restrict__ xcur) {
    int t = blockIdx.x * blockDim.x + threadIdx.x;
    int c = t & 63;
    int n = (t >> 6) & (NN - 1);
    int b = t >> 18;
    xcur[t] = x[((size_t)(b * CC + c)) * NN + n];
}

// ---------------------------------------------------------------------------
// 2) row-normalize: xhat[row] = xcur[row] / ||xcur[row]||  (one wave per row)
__global__ void k_normrows(const float* __restrict__ xcur, float* __restrict__ xhat) {
    int lane = threadIdx.x & 31;
    int row  = blockIdx.x * 4 + (threadIdx.x >> 5);
    const float* p = xcur + (size_t)row * CC;
    float v0 = p[lane], v1 = p[lane + 32];
    float s = v0 * v0 + v1 * v1;
#pragma unroll
    for (int off = 16; off > 0; off >>= 1) s += __shfl_xor(s, off, 32);
    float inv = rsqrtf(s);
    float* q = xhat + (size_t)row * CC;
    q[lane] = v0 * inv;
    q[lane + 32] = v1 * inv;
}

// ---------------------------------------------------------------------------
// 3) Gram + threshold -> bitmask (fp32 WMMA; adjacency bits must be exact)
__global__ void k_gram(const float* __restrict__ xhat, unsigned short* __restrict__ mask) {
    int lane = threadIdx.x & 31;
    int wv   = threadIdx.x >> 5;
    int j    = blockIdx.x * 8 + wv;
    int i    = blockIdx.y;
    int b    = blockIdx.z;
    int l15  = lane & 15;
    int k2   = (lane >> 4) * 2;

    const float* xb = xhat + (size_t)b * NN * CC;
    const float* pa = xb + (size_t)(i * 16 + l15) * CC + k2;
    const float* pb = xb + (size_t)(j * 16 + l15) * CC + k2;

    v8f acc = {};
#pragma unroll
    for (int k0 = 0; k0 < CC; k0 += 4) {
        v2f av = *(const v2f*)(pa + k0);
        v2f bv = *(const v2f*)(pb + k0);
        acc = wmma_f32(av, bv, acc);
    }

    unsigned bal[8];
#pragma unroll
    for (int v = 0; v < 8; ++v)
        bal[v] = __builtin_amdgcn_ballot_w32(acc[v] > THRESH);

    if (lane == 0) {
        unsigned short* mb = mask + ((size_t)b * NN + i * 16) * 256 + j;
#pragma unroll
        for (int v = 0; v < 8; ++v) {
            mb[(size_t)v * 256]       = (unsigned short)(bal[v] & 0xffffu);
            mb[(size_t)(v + 8) * 256] = (unsigned short)(bal[v] >> 16);
        }
    }
}

// ---------------------------------------------------------------------------
// 4) degree -> d_is = 1/sqrt(deg)
__global__ void k_degree(const unsigned short* __restrict__ mask, float* __restrict__ dis) {
    int row = blockIdx.x * blockDim.x + threadIdx.x;
    const unsigned* mr = (const unsigned*)(mask + (size_t)row * 256);
    int cnt = 0;
#pragma unroll 8
    for (int i = 0; i < 128; ++i) cnt += __popc(mr[i]);
    dis[row] = rsqrtf((float)cnt);
}

// ---------------------------------------------------------------------------
// 5) yT[b,c,m] = bf16( d_is[b,m] * x[b,m,c] )  (layer-0 input only)
__global__ void k_make_yT(const float* __restrict__ xin, const float* __restrict__ dis,
                          unsigned short* __restrict__ yT) {
    int t = blockIdx.x * blockDim.x + threadIdx.x;          // 0 .. B*C*N-1
    int m = t & (NN - 1);
    int c = (t >> 12) & 63;
    int b = t >> 18;
    float v = xin[(((size_t)b << 12) + m) * CC + c] * dis[(b << 12) + m];
    yT[t] = f2bf(v);
}

// ---------------------------------------------------------------------------
// 6) lx[n,:] = d_is[n] * sum_m bit(n,m) * y[m,:]  via V_WMMA_F32_16X16X32_BF16.
//    A = 0/1 adjacency expanded via 4KB LDS LUT (8 bits -> 4 packed-bf16 dwords),
//    B = yT (column-major bf16 -> 32 contiguous bytes per lane per step).
__global__ void k_spmm_bf16(const unsigned* __restrict__ mask32,
                            const unsigned short* __restrict__ yT,
                            const float* __restrict__ dis, float* __restrict__ lx) {
    __shared__ __align__(16) unsigned lut[256][4];
    int tid = threadIdx.x;
    for (int e = tid; e < 256; e += 128) {
#pragma unroll
        for (int d = 0; d < 4; ++d) {
            unsigned lo = (e >> (2 * d)) & 1u;
            unsigned hi = (e >> (2 * d + 1)) & 1u;
            lut[e][d] = lo * 0x3F80u + hi * 0x3F800000u;
        }
    }
    __syncthreads();

    int lane = tid & 31;
    int wv   = tid >> 5;                                    // 0..3 col tile
    int i16  = blockIdx.x * 16;
    int b    = blockIdx.y;
    int l15  = lane & 15;
    int g    = lane >> 4;
    int col  = wv * 16 + l15;
    int row  = i16 + l15;

    const unsigned* mrow = mask32 + ((size_t)b * NN + row) * 128;
    const __bf16*   yb   = (const __bf16*)yT + ((size_t)b * CC + col) * NN + 16 * g;

    v8f acc = {};
    for (int kq = 0; kq < 32; ++kq) {                       // 4 mask words per vmem
        v4u wq = *(const v4u*)(mrow + kq * 4);
#pragma unroll
        for (int s = 0; s < 4; ++s) {
            unsigned w = wq[s] >> (g << 4);                 // this lane-half's window
            ABpack ab;
            ab.a = *(const v4u*)lut[w & 0xFFu];             // K pairs 0..7  (VGPR 0-3)
            ab.b = *(const v4u*)lut[(w >> 8) & 0xFFu];      // K pairs 8..15 (VGPR 4-7)
            v16bf a  = __builtin_bit_cast(v16bf, ab);
            v16bf bv = *(const v16bf*)(yb + (size_t)(kq * 4 + s) * 32);
            acc = __builtin_amdgcn_wmma_f32_16x16x32_bf16(false, a, false, bv,
                                                          (short)0, acc, false, false);
        }
    }

    const float* db = dis + b * NN;
    float* lb = lx + (size_t)b * NN * CC;
    int mhi = 8 * g;
#pragma unroll
    for (int v = 0; v < 8; ++v) {
        int rm = i16 + v + mhi;
        lb[(size_t)rm * CC + col] = acc[v] * db[rm];
    }
}

// ---------------------------------------------------------------------------
// 7) h = lx @ W (fp32 WMMA) + instance_norm + relu.
//    If yT_out != null, also emit next layer's scaled bf16 column-major input
//    (fused: saves a full f32 round-trip between layers). Else write f32 out.
__global__ void k_dense_norm(const float* __restrict__ lx, const float* __restrict__ W,
                             const float* __restrict__ dis, float* __restrict__ out,
                             unsigned short* __restrict__ yT_out) {
    __shared__ float hb[16][68];
    __shared__ float mean_s[16], inv_s[16];
    int tid  = threadIdx.x;
    int lane = tid & 31;
    int wv   = tid >> 5;
    int i16  = blockIdx.x * 16;
    int b    = blockIdx.y;
    int l15  = lane & 15;
    int k2   = (lane >> 4) * 2;
    int f    = wv * 16 + l15;

    const float* lb = lx + (size_t)b * NN * CC + (size_t)(i16 + l15) * CC + k2;
    v8f acc = {};
#pragma unroll
    for (int k0 = 0; k0 < CC; k0 += 4) {
        v2f av = *(const v2f*)(lb + k0);
        v2f bv;
        bv.x = W[(size_t)(k0 + k2) * CC + f];
        bv.y = W[(size_t)(k0 + k2 + 1) * CC + f];
        acc = wmma_f32(av, bv, acc);
    }
    int mhi = 8 * (lane >> 4);
#pragma unroll
    for (int v = 0; v < 8; ++v) hb[v + mhi][f] = acc[v];
    __syncthreads();

    if (tid < 16) {
        float s = 0.f;
#pragma unroll
        for (int c = 0; c < CC; ++c) s += hb[tid][c];
        float m = s * (1.0f / 64.0f);
        float vs = 0.f;
#pragma unroll
        for (int c = 0; c < CC; ++c) { float d = hb[tid][c] - m; vs += d * d; }
        mean_s[tid] = m;
        inv_s[tid]  = rsqrtf(vs * (1.0f / 64.0f) + EPSN);
    }
    __syncthreads();

    if (yT_out) {
        for (int idx = tid; idx < 16 * CC; idx += 128) {
            int r = idx >> 6, c = idx & 63;
            int m = i16 + r;
            float val = (hb[r][c] - mean_s[r]) * inv_s[r];
            val = val > 0.f ? val : 0.f;
            yT_out[((size_t)b * CC + c) * NN + m] = f2bf(val * dis[b * NN + m]);
        }
    } else {
        float* ob = out + ((size_t)b * NN + i16) * CC;
        for (int idx = tid; idx < 16 * CC; idx += 128) {
            int r = idx >> 6, c = idx & 63;
            float val = (hb[r][c] - mean_s[r]) * inv_s[r];
            ob[(size_t)r * CC + c] = val > 0.f ? val : 0.f;
        }
    }
}

// ---------------------------------------------------------------------------
extern "C" void kernel_launch(void* const* d_in, const int* in_sizes, int n_in,
                              void* d_out, int out_size, void* d_ws, size_t ws_size,
                              hipStream_t stream) {
    (void)in_sizes; (void)n_in; (void)out_size; (void)ws_size;
    const float* x  = (const float*)d_in[0];   // [4,64,64,64]
    const float* W0 = (const float*)d_in[1];   // [1,64,64]
    const float* W1 = (const float*)d_in[2];   // [1,64,64]
    float* out = (float*)d_out;                // [4,4096,64]

    char* ws = (char*)d_ws;
    float*          xcur = (float*)(ws + (size_t)0);                        // 4 MB
    float*          xhat = (float*)(ws + (size_t)4  * 1024 * 1024);         // 4 MB
    unsigned short* yT   = (unsigned short*)(ws + (size_t)8 * 1024 * 1024); // 2 MB
    float*          lxb  = (float*)(ws + (size_t)12 * 1024 * 1024);         // 4 MB
    float*          dis  = (float*)(ws + (size_t)16 * 1024 * 1024);         // 64 KB
    unsigned short* mask = (unsigned short*)(ws + (size_t)17 * 1024 * 1024);// 8 MB

    // features + adjacency bitmask + degree scaling (built once, as in reference)
    k_transpose<<<dim3(BN * NN * CC / 256), 256, 0, stream>>>(x, xcur);
    k_normrows <<<dim3(BN * NN / 4), 128, 0, stream>>>(xcur, xhat);
    k_gram     <<<dim3(32, 256, BN), 256, 0, stream>>>(xhat, mask);
    k_degree   <<<dim3(BN * NN / 256), 256, 0, stream>>>(mask, dis);

    // layer 0 (dense epilogue emits layer-1's bf16 input directly)
    k_make_yT   <<<dim3(BN * NN * CC / 256), 256, 0, stream>>>(xcur, dis, yT);
    k_spmm_bf16 <<<dim3(256, BN), 128, 0, stream>>>((const unsigned*)mask, yT, dis, lxb);
    k_dense_norm<<<dim3(256, BN), 128, 0, stream>>>(lxb, W0, dis, nullptr, yT);

    // layer 1 (adjacency reused)
    k_spmm_bf16 <<<dim3(256, BN), 128, 0, stream>>>((const unsigned*)mask, yT, dis, lxb);
    k_dense_norm<<<dim3(256, BN), 128, 0, stream>>>(lxb, W1, dis, out, nullptr);
}
